// fkopt_net_17454747091097
// MI455X (gfx1250) — matrix operand order repaired and checked by reading, not compiled, not run on hardware
//
#include <hip/hip_runtime.h>
#include <hip/hip_bf16.h>
#include <math.h>

typedef __attribute__((ext_vector_type(16))) _Float16 v16h;
typedef __attribute__((ext_vector_type(8)))  float    v8f;

#define Bsz   65536
#define Hdim  1024
#define Jn    24
#define MROWS 32      // batch rows per block (LDS = 129 KB -> safe under any CU-mode cap)
#define KS    1032    // LDS activation row stride in halves (1024 + 8 pad -> 4-bank rotate/row)
#define QS    100     // q row stride in floats
#define NTC   8       // N-tiles per chunk: each A fragment feeds 8 WMMAs

union AF { v16h v; uint4 u[2]; };

// ---------- prep: fp32 (K,N) row-major -> f16 transposed (N,K) row-major ----------
__global__ void cvt_transpose(const float* __restrict__ W, _Float16* __restrict__ Wt,
                              int K, int N) {
  int idx = blockIdx.x * blockDim.x + threadIdx.x;
  if (idx >= K * N) return;
  int k = idx / N, n = idx - k * N;
  Wt[n * K + k] = (_Float16)W[idx];
}

// ---------- one 32x1024x1024 GEMM layer: src(LDS f16) x Wt(L2 f16,transposed) + bias, relu -> dst(LDS f16)
__device__ __forceinline__ void gemm_relu_layer(const _Float16* src,
                                                const _Float16* __restrict__ Wt,
                                                const float* __restrict__ bias,
                                                _Float16* dst, int wave, int lane) {
  const int mt    = wave & 1;           // M tile (16 rows) owned by this wave
  const int nbase = (wave >> 1) * 16;   // first of 16 N tiles owned by this wave
  const int lg    = (lane >> 4) & 1;    // lane group
  const int ln    = lane & 15;
  const _Float16* arow = src + (mt * 16 + ln) * KS;

  for (int ntc = 0; ntc < 16 / NTC; ++ntc) {
    v8f acc[NTC] = {};
    for (int kt = 0; kt < 32; ++kt) {
      AF a;
      const _Float16* ap = arow + kt * 32 + lg * 8;
      a.u[0] = *(const uint4*)ap;            // K = kt*32 + lg*8 + 0..7
      a.u[1] = *(const uint4*)(ap + 16);     // K = kt*32 + lg*8 + 16..23
#pragma unroll
      for (int i = 0; i < NTC; ++i) {
        int n0 = (nbase + ntc * NTC + i) * 16;
        const _Float16* bp = Wt + (n0 + ln) * Hdim + kt * 32 + lg * 8;
        AF b;
        b.u[0] = *(const uint4*)bp;          // K = kt*32 + lg*8 + 0..7
        b.u[1] = *(const uint4*)(bp + 16);   // K = kt*32 + lg*8 + 16..23
        acc[i] = __builtin_amdgcn_wmma_f32_16x16x32_f16(
            false, a.v, false, b.v, (short)0, acc[i], false, false);
      }
    }
#pragma unroll
    for (int i = 0; i < NTC; ++i) {
      int n0 = (nbase + ntc * NTC + i) * 16;
      float bv = bias[n0 + ln];
#pragma unroll
      for (int r = 0; r < 8; ++r) {
        float v = fmaxf(acc[i][r] + bv, 0.0f);
        dst[(mt * 16 + lg * 8 + r) * KS + n0 + ln] = (_Float16)v;
      }
    }
  }
}

// ---------- fused MLP + quaternion FK ----------
__global__ void __launch_bounds__(256) fk_fused(
    const float* __restrict__ x,
    const float* __restrict__ W1, const float* __restrict__ b1,
    const _Float16* __restrict__ Wt2, const float* __restrict__ b2,
    const _Float16* __restrict__ Wt3, const float* __restrict__ b3,
    const _Float16* __restrict__ Wt4, const float* __restrict__ b4,
    float* __restrict__ out) {
  extern __shared__ _Float16 smem[];
  _Float16* hA = smem;
  _Float16* hB = smem + MROWS * KS;
  float*    q  = (float*)hB;            // layer-4 output reuses hB (32 x 96 f32)

  const int tid  = threadIdx.x;
  const int wave = tid >> 5;
  const int lane = tid & 31;
  const int row0 = blockIdx.x * MROWS;

  // ---- layer 1: h1 = relu(x @ W1 + b1), K=3 -> plain VALU into LDS f16
  for (int idx = tid; idx < MROWS * Hdim; idx += 256) {
    int r = idx >> 10, c = idx & (Hdim - 1);
    const float* xr = x + (row0 + r) * 3;
    float v = xr[0] * W1[c] + xr[1] * W1[Hdim + c] + xr[2] * W1[2 * Hdim + c] + b1[c];
    hA[r * KS + c] = (_Float16)fmaxf(v, 0.0f);
  }
  __syncthreads();

  // ---- layers 2 & 3: WMMA GEMMs
  gemm_relu_layer(hA, Wt2, b2, hB, wave, lane);
  __syncthreads();
  gemm_relu_layer(hB, Wt3, b3, hA, wave, lane);
  __syncthreads();

  // ---- layer 4: 32x96 = 12 WMMA tiles strided across 8 waves, tanh, f32 -> q
  {
    const int lg = (lane >> 4) & 1;
    const int ln = lane & 15;
    for (int t = wave; t < 12; t += 8) {
      const int mt = t / 6;
      const int nt = t % 6;
      const int n0 = nt * 16;
      const _Float16* arow = hA + (mt * 16 + ln) * KS;
      v8f acc = {};
      for (int kt = 0; kt < 32; ++kt) {
        AF a;
        const _Float16* ap = arow + kt * 32 + lg * 8;
        a.u[0] = *(const uint4*)ap;
        a.u[1] = *(const uint4*)(ap + 16);
        const _Float16* bp = Wt4 + (n0 + ln) * Hdim + kt * 32 + lg * 8;
        AF b;
        b.u[0] = *(const uint4*)bp;
        b.u[1] = *(const uint4*)(bp + 16);
        acc = __builtin_amdgcn_wmma_f32_16x16x32_f16(
            false, a.v, false, b.v, (short)0, acc, false, false);
      }
      float bv = b4[n0 + ln];
#pragma unroll
      for (int r = 0; r < 8; ++r)
        q[(mt * 16 + lg * 8 + r) * QS + n0 + ln] = tanhf(acc[r] + bv);
    }
  }
  __syncthreads();

  // ---- FK scan: 24-joint chain per row; coordi = [R | 0.2*R[:,1]; 1 0 0 1]
  if (tid < MROWS) {
    const float* qr = q + tid * QS;
    float carry[16] = {1,0,0,0, 0,1,0,0, 0,0,1,0, 0,0,0,1};
    float* o = out + (size_t)(row0 + tid) * (Jn * 3);
    for (int j = 0; j < Jn; ++j) {
      float a0 = qr[j*4+0], a1 = qr[j*4+1], a2 = qr[j*4+2], a3 = qr[j*4+3];
      float s = 2.0f / (a0*a0 + a1*a1 + a2*a2 + a3*a3);
      float R00 = 1.0f - s*(a2*a2 + a3*a3);
      float R01 = s*(a1*a2 - a3*a0);
      float R02 = s*(a1*a3 + a2*a0);
      float R10 = s*(a1*a2 + a3*a0);
      float R11 = 1.0f - s*(a1*a1 + a3*a3);
      float R12 = s*(a2*a3 - a1*a0);
      float R20 = s*(a1*a3 - a2*a0);
      float R21 = s*(a2*a3 + a1*a0);
      float R22 = 1.0f - s*(a1*a1 + a2*a2);
      float co[16] = { R00, R01, R02, 0.2f*R01,
                       R10, R11, R12, 0.2f*R11,
                       R20, R21, R22, 0.2f*R21,
                       1.0f, 0.0f, 0.0f, 1.0f };
      float M[16];
#pragma unroll
      for (int i2 = 0; i2 < 4; ++i2)
#pragma unroll
        for (int l2 = 0; l2 < 4; ++l2)
          M[i2*4+l2] = co[i2*4+0]*carry[l2]     + co[i2*4+1]*carry[4+l2]
                     + co[i2*4+2]*carry[8+l2]   + co[i2*4+3]*carry[12+l2];
#pragma unroll
      for (int t2 = 0; t2 < 16; ++t2) carry[t2] = M[t2];
      float inv = 1.0f / (M[15] + 1e-9f);
      o[j*3+0] = M[3]  * inv;
      o[j*3+1] = M[7]  * inv;
      o[j*3+2] = M[11] * inv;
    }
  }
}

extern "C" void kernel_launch(void* const* d_in, const int* in_sizes, int n_in,
                              void* d_out, int out_size, void* d_ws, size_t ws_size,
                              hipStream_t stream) {
  (void)in_sizes; (void)n_in; (void)out_size; (void)ws_size;
  const float* x  = (const float*)d_in[0];
  const float* W1 = (const float*)d_in[1];
  const float* b1 = (const float*)d_in[2];
  const float* W2 = (const float*)d_in[3];
  const float* b2 = (const float*)d_in[4];
  const float* W3 = (const float*)d_in[5];
  const float* b3 = (const float*)d_in[6];
  const float* W4 = (const float*)d_in[7];
  const float* b4 = (const float*)d_in[8];
  float* out = (float*)d_out;

  char* ws = (char*)d_ws;
  _Float16* Wt2 = (_Float16*)ws;                                  // 2 MB
  _Float16* Wt3 = (_Float16*)(ws + (size_t)Hdim * Hdim * 2);      // 2 MB
  _Float16* Wt4 = (_Float16*)(ws + (size_t)Hdim * Hdim * 4);      // 192 KB

  cvt_transpose<<<(Hdim * Hdim + 255) / 256, 256, 0, stream>>>(W2, Wt2, Hdim, Hdim);
  cvt_transpose<<<(Hdim * Hdim + 255) / 256, 256, 0, stream>>>(W3, Wt3, Hdim, Hdim);
  cvt_transpose<<<(Hdim * 96   + 255) / 256, 256, 0, stream>>>(W4, Wt4, Hdim, 96);

  size_t smem = (size_t)2 * MROWS * KS * sizeof(_Float16);        // 129 KB
  fk_fused<<<Bsz / MROWS, 256, smem, stream>>>(x, W1, b1, Wt2, b2, Wt3, b3, Wt4, b4, out);
}
